// GATNet_34359738928
// MI455X (gfx1250) — matrix operand order repaired and checked by reading, not compiled, not run on hardware
//
#include <hip/hip_runtime.h>
#include <stdint.h>

// ---------------- WMMA / TDM vector types (CDNA5 / gfx1250, wave32) ---------
typedef __attribute__((ext_vector_type(16))) __bf16    v16bf;
typedef __attribute__((ext_vector_type(8)))  float     v8f;
typedef __attribute__((ext_vector_type(4)))  unsigned  uint4v;
typedef __attribute__((ext_vector_type(8)))  int       int8v;
typedef __attribute__((ext_vector_type(4)))  int       int4v;

#define F_IN  500
#define KPAD  512
#define WLSTR 520   // LDS stride (bf16) for W tile (bank-conflict padding)
#define XLSTR 34    // LDS stride (f32/bf16) for X tile: 32 data + 2 pad DWORDs

// ---- Tensor Data Mover availability / arity selection ----------------------
#ifdef __has_builtin
# if __has_builtin(__builtin_amdgcn_tensor_load_to_lds)
#  define HAVE_TDM 1
# endif
#endif
#ifndef HAVE_TDM
# define HAVE_TDM 0
#endif

#if HAVE_TDM
# if __has_include(<hip/amd_detail/amd_gfx1250_TDM.h>)
   // therock-10.0 headers -> clang-23 6-arg builtin
#  define TDM_LOAD(g0, g1, g2, g3) \
      __builtin_amdgcn_tensor_load_to_lds((g0), (g1), (g2), (g3), (int8v)0, 0)
# else
   // ROCm 7.2 clang-22 5-arg builtin
#  define TDM_LOAD(g0, g1, g2, g3) \
      __builtin_amdgcn_tensor_load_to_lds((g0), (g1), (g2), (g3), 0)
# endif

__device__ __forceinline__ void tdm_wait0() {
# if __has_builtin(__builtin_amdgcn_s_wait_tensorcnt)
    __builtin_amdgcn_s_wait_tensorcnt((short)0);
# else
    asm volatile("s_wait_tensorcnt 0x0" ::: "memory");
# endif
}

// Issue a 2-D TDM load: tile [128 rows x 32 f32 cols] from x (row stride 500),
// OOB (cols >= remCols, rows >= remRows) zero-filled by hardware; LDS row
// stride padded to 34 DWORDs via pad_interval=32DW / pad_amount=2DW.
__device__ __forceinline__ void tdm_load_tile(const float* gsrc, unsigned ldsOff,
                                              int remCols, int remRows) {
    unsigned long long ga = (unsigned long long)(uintptr_t)gsrc;
    uint4v g0;
    g0[0] = 1u;                                            // count=1, user desc
    g0[1] = ldsOff;                                        // lds_addr (bytes)
    g0[2] = (unsigned)(ga & 0xFFFFFFFFu);                  // global_addr[31:0]
    g0[3] = (unsigned)((ga >> 32) & 0x1FFFFFFu)            // global_addr[56:32]
          | 0x80000000u;                                   // type=2 (bits 127:126)
    int8v g1;
    g1[0] = (int)((2u << 16)        // data_size = 4 bytes
                | (1u << 20)        // pad_enable
                | (4u << 22)        // pad_interval: 2^4*2 = 32 DWORDs
                | (1u << 25));      // pad_amount: 2 DWORDs
    unsigned td0 = (unsigned)remCols, td1 = (unsigned)remRows;
    g1[1] = (int)((td0 & 0xFFFFu) << 16);                  // tensor_dim0[15:0]
    g1[2] = (int)(((td0 >> 16) & 0xFFFFu) | ((td1 & 0xFFFFu) << 16));
    g1[3] = (int)(((td1 >> 16) & 0xFFFFu) | (32u << 16));  // tile_dim0 = 32
    g1[4] = (int)128;                                      // tile_dim1=128, tile_dim2=0
    g1[5] = (int)F_IN;                                     // tensor_dim0_stride = 500
    g1[6] = 0;
    g1[7] = 0;
    int4v z = {};
    TDM_LOAD(g0, g1, z, z);
}
#endif // HAVE_TDM

__device__ __forceinline__ float leaky(float x) { return x > 0.f ? x : 0.2f * x; }

// Order-preserving float -> uint key for atomicMax over signed floats.
__device__ __forceinline__ unsigned fkey(float f) {
    unsigned u = __float_as_uint(f);
    return (u & 0x80000000u) ? ~u : (u | 0x80000000u);
}
__device__ __forceinline__ float funkey(unsigned k) {
    unsigned u = (k & 0x80000000u) ? (k & 0x7FFFFFFFu) : ~k;
    return __uint_as_float(u);
}
#define KEY_NEG_INF 0x007FFFFFu   // fkey(-inf)

// ---------------------------------------------------------------------------
// GEMM1: h1[N,64] = x[N,500] @ W1[500,64], bf16 WMMA, f32 accumulate.
// Block = 256 thr (8 waves). Wave -> 16x64 C tile; block -> 128 rows.
// X tiles are DMA'd by the Tensor Data Mover (double-buffered, f32 in LDS),
// converted to bf16 in registers while building A fragments.
// ---------------------------------------------------------------------------
__global__ __launch_bounds__(256) void k_gemm1(const float* __restrict__ x,
                                               const float* __restrict__ W1,
                                               float* __restrict__ h1, int N) {
    __shared__ alignas(16) __bf16 Wl[64 * WLSTR];    // [col][k] transposed, ~66.5 KB
#if HAVE_TDM
    __shared__ alignas(16) float  Xf[2][128 * XLSTR]; // double-buffered f32 tiles, 34 KB
#else
    __shared__ alignas(16) __bf16 Xl[128 * XLSTR];    // manual-staged bf16 tile
#endif

    const int tid = threadIdx.x;
    const int rowBase = blockIdx.x * 128;

    // Stage W1 -> LDS (transposed, bf16, zero-padded K to 512). Once per block.
    for (int idx = tid; idx < 64 * KPAD; idx += 256) {
        int k = idx >> 6, c = idx & 63;
        float v = (k < F_IN) ? W1[k * 64 + c] : 0.f;
        Wl[c * WLSTR + k] = (__bf16)v;
    }

    v8f acc[4] = {};   // 4 column-groups of 16

    const int lane    = tid & 31;
    const int w       = tid >> 5;
    const int halfSel = (lane < 16) ? 0 : 1;           // K-half select (ISA A/B layout)
    const int mloc    = w * 16 + (lane & 15);          // A-fragment row

#if HAVE_TDM
    const unsigned lds0 = (unsigned)(uintptr_t)&Xf[0][0];
    const unsigned lds1 = (unsigned)(uintptr_t)&Xf[1][0];
    const int remRows = N - rowBase;
    // prologue: DMA tile 0 into buffer 0
    if (w == 0) {
        tdm_load_tile(x + (long)rowBase * F_IN, lds0, F_IN, remRows);
        tdm_wait0();
    }
    __syncthreads();

    for (int kt = 0; kt < KPAD / 32; ++kt) {
        const int k0 = kt * 32;
        // overlap: DMA next tile into the other buffer while computing
        if (w == 0 && kt + 1 < KPAD / 32) {
            int k1 = k0 + 32;
            tdm_load_tile(x + (long)rowBase * F_IN + k1,
                          ((kt + 1) & 1) ? lds1 : lds0, F_IN - k1, remRows);
        }
        // ---- A fragment: read f32 from LDS, convert to bf16 in registers
        v16bf a;
        {
            const float* xr = &Xf[kt & 1][mloc * XLSTR + halfSel * 8];
#pragma unroll
            for (int v = 0; v < 4; ++v) {
                a[2 * v]         = (__bf16)xr[2 * v];
                a[2 * v + 1]     = (__bf16)xr[2 * v + 1];
                a[8 + 2 * v]     = (__bf16)xr[16 + 2 * v];
                a[8 + 2 * v + 1] = (__bf16)xr[16 + 2 * v + 1];
            }
        }
        // ---- B fragments (32x16 bf16): lanes<16 K=0..7,16..23, lanes>=16 K=8..15,24..31
#pragma unroll
        for (int cg = 0; cg < 4; ++cg) {
            const int col = cg * 16 + (lane & 15);
            const __bf16* wr = &Wl[col * WLSTR + k0 + halfSel * 8];
            v16bf b;
#pragma unroll
            for (int v = 0; v < 4; ++v) {
                b[2 * v]         = wr[2 * v];
                b[2 * v + 1]     = wr[2 * v + 1];
                b[8 + 2 * v]     = wr[16 + 2 * v];
                b[8 + 2 * v + 1] = wr[16 + 2 * v + 1];
            }
            acc[cg] = __builtin_amdgcn_wmma_f32_16x16x32_bf16(
                false, a, false, b, (short)0, acc[cg], false, false);
        }
        __syncthreads();                          // all waves done reading buf[kt&1]
        if (w == 0 && kt + 1 < KPAD / 32) tdm_wait0();  // next tile landed
        __syncthreads();                          // visibility for all waves
    }
#else
    __syncthreads();
    for (int kt = 0; kt < KPAD / 32; ++kt) {
        const int k0 = kt * 32;
        {   // manual staging fallback
            int rl  = tid >> 1;
            int kh  = (tid & 1) * 16;
            int row = rowBase + rl;
            __bf16* dst = &Xl[rl * XLSTR + kh];
            if (row < N) {
                const float* srcp = &x[(long)row * F_IN + k0 + kh];
                if (k0 + kh + 16 <= F_IN) {
#pragma unroll
                    for (int j = 0; j < 16; ++j) dst[j] = (__bf16)srcp[j];
                } else {
#pragma unroll
                    for (int j = 0; j < 16; ++j) {
                        int k = k0 + kh + j;
                        dst[j] = (k < F_IN) ? (__bf16)srcp[j] : (__bf16)0.f;
                    }
                }
            } else {
#pragma unroll
                for (int j = 0; j < 16; ++j) dst[j] = (__bf16)0.f;
            }
        }
        __syncthreads();
        v16bf a;
        {
            const __bf16* xr = &Xl[mloc * XLSTR + halfSel * 8];
#pragma unroll
            for (int v = 0; v < 4; ++v) {
                a[2 * v]         = xr[2 * v];
                a[2 * v + 1]     = xr[2 * v + 1];
                a[8 + 2 * v]     = xr[16 + 2 * v];
                a[8 + 2 * v + 1] = xr[16 + 2 * v + 1];
            }
        }
#pragma unroll
        for (int cg = 0; cg < 4; ++cg) {
            const int col = cg * 16 + (lane & 15);
            const __bf16* wr = &Wl[col * WLSTR + k0 + halfSel * 8];
            v16bf b;
#pragma unroll
            for (int v = 0; v < 4; ++v) {
                b[2 * v]         = wr[2 * v];
                b[2 * v + 1]     = wr[2 * v + 1];
                b[8 + 2 * v]     = wr[16 + 2 * v];
                b[8 + 2 * v + 1] = wr[16 + 2 * v + 1];
            }
            acc[cg] = __builtin_amdgcn_wmma_f32_16x16x32_bf16(
                false, a, false, b, (short)0, acc[cg], false, false);
        }
        __syncthreads();
    }
#endif

    // ---- epilogue: C layout VGPR i -> M = i (lanes<16) / 8+i (lanes>=16), N = lane&15
#pragma unroll
    for (int cg = 0; cg < 4; ++cg) {
#pragma unroll
        for (int i = 0; i < 8; ++i) {
            int row = rowBase + w * 16 + (halfSel ? 8 + i : i);
            int col = cg * 16 + (lane & 15);
            if (row < N) h1[(long)row * 64 + col] = acc[cg][i];
        }
    }
}

// ---------------- attention coefficients layer 1 ---------------------------
__global__ void k_attn1(const float* __restrict__ h1, const float* __restrict__ as,
                        const float* __restrict__ ad, float* a1s, float* a1d, int N) {
    int t = blockIdx.x * blockDim.x + threadIdx.x;
    if (t >= N * 8) return;
    int n = t >> 3, h = t & 7;
    const float* hp = &h1[(long)n * 64 + h * 8];
    float ss = 0.f, sd = 0.f;
#pragma unroll
    for (int c = 0; c < 8; ++c) { ss += hp[c] * as[h * 8 + c]; sd += hp[c] * ad[h * 8 + c]; }
    a1s[t] = ss; a1d[t] = sd;
}

__global__ void k_fill_u32(unsigned* p, unsigned v, int n) {
    int t = blockIdx.x * blockDim.x + threadIdx.x;
    if (t < n) p[t] = v;
}

// ---------------- layer-1 edge passes (H = 8) ------------------------------
__global__ void k_edge_max1(const long long* __restrict__ ei, int E,
                            const float* __restrict__ a1s, const float* __restrict__ a1d,
                            unsigned* __restrict__ m1) {
    int e = blockIdx.x * blockDim.x + threadIdx.x;
    if (e >= E) return;
    int s = (int)ei[e], d = (int)ei[(long)E + e];
#pragma unroll
    for (int h = 0; h < 8; ++h) {
        float l = leaky(a1s[s * 8 + h] + a1d[d * 8 + h]);
        atomicMax(&m1[d * 8 + h], fkey(l));
    }
}

__global__ void k_decode1(const float* a1s, const float* a1d, float* m1f, float* den1, int N8) {
    int t = blockIdx.x * blockDim.x + threadIdx.x;
    if (t >= N8) return;
    float lv = leaky(a1s[t] + a1d[t]);               // self-loop logit
    float me = funkey(((const unsigned*)m1f)[t]);
    float m  = (me >= -3.0e38f) ? fmaxf(me, lv) : lv;
    m1f[t]  = m;
    den1[t] = __expf(lv - m);                        // seed denom with loop term
}

__global__ void k_edge_sum1(const long long* __restrict__ ei, int E,
                            const float* a1s, const float* a1d,
                            const float* m1f, float* den1) {
    int e = blockIdx.x * blockDim.x + threadIdx.x;
    if (e >= E) return;
    int s = (int)ei[e], d = (int)ei[(long)E + e];
#pragma unroll
    for (int h = 0; h < 8; ++h) {
        float l = leaky(a1s[s * 8 + h] + a1d[d * 8 + h]);
        atomicAdd(&den1[d * 8 + h], __expf(l - m1f[d * 8 + h]));
    }
}

__global__ void k_loopmsg1(const float* h1, const float* a1s, const float* a1d,
                           const float* m1f, const float* den1, float* out1, int N64) {
    int t = blockIdx.x * blockDim.x + threadIdx.x;
    if (t >= N64) return;
    int n = t >> 6, ch = t & 63, h = ch >> 3;
    float lv = leaky(a1s[n * 8 + h] + a1d[n * 8 + h]);
    float alpha = __expf(lv - m1f[n * 8 + h]) / (den1[n * 8 + h] + 1e-16f);
    out1[t] = h1[t] * alpha;                         // init + self-loop message
}

// wave-per-edge: 32 lanes sweep 64 channels -> coalesced gather + L2 atomics
__global__ void k_edge_msg1(const long long* __restrict__ ei, int E,
                            const float* a1s, const float* a1d,
                            const float* m1f, const float* den1,
                            const float* __restrict__ h1, float* __restrict__ out1) {
    long g = blockIdx.x * (long)blockDim.x + threadIdx.x;
    long e = g >> 5; int lane = (int)(g & 31);
    if (e >= E) return;
    int s = (int)ei[e], d = (int)ei[(long)E + e];
#pragma unroll
    for (int r = 0; r < 2; ++r) {
        int ch = r * 32 + lane, h = ch >> 3;
        float l = leaky(a1s[s * 8 + h] + a1d[d * 8 + h]);
        float alpha = __expf(l - m1f[d * 8 + h]) / (den1[d * 8 + h] + 1e-16f);
        atomicAdd(&out1[(long)d * 64 + ch], h1[(long)s * 64 + ch] * alpha);
    }
}

// ---------------- layer 2: tiny GEMM + attention (fused per node) ----------
__global__ void k_layer2(const float* out1, const float* b1, const float* W2,
                         const float* at2s, const float* at2d,
                         float* h2, float* a2s, float* a2d, int N) {
    int n = blockIdx.x * blockDim.x + threadIdx.x;
    if (n >= N) return;
    float acc[7] = {0, 0, 0, 0, 0, 0, 0};
    for (int ch = 0; ch < 64; ++ch) {
        float v = out1[(long)n * 64 + ch] + b1[ch];
#pragma unroll
        for (int j = 0; j < 7; ++j) acc[j] += v * W2[ch * 7 + j];
    }
    float ss = 0.f, sd = 0.f;
#pragma unroll
    for (int j = 0; j < 7; ++j) {
        h2[n * 7 + j] = acc[j];
        ss += acc[j] * at2s[j];
        sd += acc[j] * at2d[j];
    }
    a2s[n] = ss; a2d[n] = sd;
}

// ---------------- layer-2 edge passes (H = 1, C = 7) -----------------------
__global__ void k_edge_max2(const long long* ei, int E, const float* a2s,
                            const float* a2d, unsigned* m2) {
    int e = blockIdx.x * blockDim.x + threadIdx.x;
    if (e >= E) return;
    int s = (int)ei[e], d = (int)ei[(long)E + e];
    atomicMax(&m2[d], fkey(leaky(a2s[s] + a2d[d])));
}

__global__ void k_decode2(const float* a2s, const float* a2d, float* m2f, float* den2, int N) {
    int n = blockIdx.x * blockDim.x + threadIdx.x;
    if (n >= N) return;
    float lv = leaky(a2s[n] + a2d[n]);
    float me = funkey(((const unsigned*)m2f)[n]);
    float m  = (me >= -3.0e38f) ? fmaxf(me, lv) : lv;
    m2f[n]  = m;
    den2[n] = __expf(lv - m);
}

__global__ void k_edge_sum2(const long long* ei, int E, const float* a2s,
                            const float* a2d, const float* m2f, float* den2) {
    int e = blockIdx.x * blockDim.x + threadIdx.x;
    if (e >= E) return;
    int s = (int)ei[e], d = (int)ei[(long)E + e];
    atomicAdd(&den2[d], __expf(leaky(a2s[s] + a2d[d]) - m2f[d]));
}

__global__ void k_loopmsg2(const float* h2, const float* a2s, const float* a2d,
                           const float* m2f, const float* den2, float* out2, int N) {
    int n = blockIdx.x * blockDim.x + threadIdx.x;
    if (n >= N) return;
    float lv = leaky(a2s[n] + a2d[n]);
    float alpha = __expf(lv - m2f[n]) / (den2[n] + 1e-16f);
#pragma unroll
    for (int j = 0; j < 7; ++j) out2[n * 7 + j] = h2[n * 7 + j] * alpha;
}

__global__ void k_edge_msg2(const long long* ei, int E, const float* a2s,
                            const float* a2d, const float* m2f, const float* den2,
                            const float* h2, float* out2) {
    int e = blockIdx.x * blockDim.x + threadIdx.x;
    if (e >= E) return;
    int s = (int)ei[e], d = (int)ei[(long)E + e];
    float alpha = __expf(leaky(a2s[s] + a2d[d]) - m2f[d]) / (den2[d] + 1e-16f);
#pragma unroll
    for (int j = 0; j < 7; ++j)
        atomicAdd(&out2[d * 7 + j], h2[s * 7 + j] * alpha);
}

// ---------------- bias + log_softmax ---------------------------------------
__global__ void k_final(const float* out2, const float* b2, float* y, int N) {
    int n = blockIdx.x * blockDim.x + threadIdx.x;
    if (n >= N) return;
    float v[7];
    float mx = -3.4e38f;
#pragma unroll
    for (int j = 0; j < 7; ++j) { v[j] = out2[n * 7 + j] + b2[j]; mx = fmaxf(mx, v[j]); }
    float s = 0.f;
#pragma unroll
    for (int j = 0; j < 7; ++j) s += __expf(v[j] - mx);
    float lse = __logf(s);
#pragma unroll
    for (int j = 0; j < 7; ++j) y[n * 7 + j] = v[j] - mx - lse;
}

// ---------------------------------------------------------------------------
extern "C" void kernel_launch(void* const* d_in, const int* in_sizes, int n_in,
                              void* d_out, int out_size, void* d_ws, size_t ws_size,
                              hipStream_t stream) {
    const float*     x    = (const float*)d_in[0];
    const long long* ei   = (const long long*)d_in[1];   // int64 [2,E]
    const float*     W1   = (const float*)d_in[2];
    const float*     at1s = (const float*)d_in[3];
    const float*     at1d = (const float*)d_in[4];
    const float*     b1   = (const float*)d_in[5];
    const float*     W2   = (const float*)d_in[6];
    const float*     at2s = (const float*)d_in[7];
    const float*     at2d = (const float*)d_in[8];
    const float*     b2   = (const float*)d_in[9];
    float*           y    = (float*)d_out;

    const int N = in_sizes[0] / F_IN;
    const int E = in_sizes[1] / 2;

    // workspace carve-out (floats), ~72 MB total
    float* ws   = (float*)d_ws;
    float* h1   = ws;                    // 64N
    float* a1s  = h1 + 64L * N;          // 8N
    float* a1d  = a1s + 8L * N;          // 8N
    float* m1   = a1d + 8L * N;          // 8N (uint keys -> float max, in place)
    float* den1 = m1 + 8L * N;           // 8N
    float* out1 = den1 + 8L * N;         // 64N
    float* h2   = out1 + 64L * N;        // 8N (7N used)
    float* a2s  = h2 + 8L * N;           // N
    float* a2d  = a2s + (long)N;         // N
    float* m2   = a2d + (long)N;         // N
    float* den2 = m2 + (long)N;          // N
    float* out2 = den2 + (long)N;        // 7N

    const int B = 256;
    auto cdiv = [](long a, long b) { return (unsigned)((a + b - 1) / b); };

    // ---- layer 1
    k_gemm1<<<cdiv(N, 128), B, 0, stream>>>(x, W1, h1, N);
    k_attn1<<<cdiv(8L * N, B), B, 0, stream>>>(h1, at1s, at1d, a1s, a1d, N);
    k_fill_u32<<<cdiv(8L * N, B), B, 0, stream>>>((unsigned*)m1, KEY_NEG_INF, 8 * N);
    k_edge_max1<<<cdiv(E, B), B, 0, stream>>>(ei, E, a1s, a1d, (unsigned*)m1);
    k_decode1<<<cdiv(8L * N, B), B, 0, stream>>>(a1s, a1d, m1, den1, 8 * N);
    k_edge_sum1<<<cdiv(E, B), B, 0, stream>>>(ei, E, a1s, a1d, m1, den1);
    k_loopmsg1<<<cdiv(64L * N, B), B, 0, stream>>>(h1, a1s, a1d, m1, den1, out1, 64 * N);
    k_edge_msg1<<<cdiv(32L * E, B), B, 0, stream>>>(ei, E, a1s, a1d, m1, den1, h1, out1);

    // ---- layer 2
    k_layer2<<<cdiv(N, B), B, 0, stream>>>(out1, b1, W2, at2s, at2d, h2, a2s, a2d, N);
    k_fill_u32<<<cdiv(N, B), B, 0, stream>>>((unsigned*)m2, KEY_NEG_INF, N);
    k_edge_max2<<<cdiv(E, B), B, 0, stream>>>(ei, E, a2s, a2d, (unsigned*)m2);
    k_decode2<<<cdiv(N, B), B, 0, stream>>>(a2s, a2d, m2, den2, N);
    k_edge_sum2<<<cdiv(E, B), B, 0, stream>>>(ei, E, a2s, a2d, m2, den2);
    k_loopmsg2<<<cdiv(N, B), B, 0, stream>>>(h2, a2s, a2d, m2, den2, out2, N);
    k_edge_msg2<<<cdiv(E, B), B, 0, stream>>>(ei, E, a2s, a2d, m2, den2, h2, out2);

    // ---- output
    k_final<<<cdiv(N, B), B, 0, stream>>>(out2, b2, y, N);
}
